// DialogueGCN_91216515433183
// MI455X (gfx1250) — matrix operand order repaired and checked by reading, not compiled, not run on hardware
//
#include <hip/hip_runtime.h>
#include <hip/hip_bf16.h>

#define N_UTT 4096
#define D_EMB 512
#define W_ATT 10
#define WIN 21
#define NREL 5                 // 4 relation weights + aggr weight fused into one GEMM
#define HWCOLS (NREL * D_EMB)  // 2560

typedef __attribute__((ext_vector_type(8)))  __bf16 v8bf;
typedef __attribute__((ext_vector_type(16))) __bf16 v16bf;
typedef __attribute__((ext_vector_type(8)))  float  v8f;

__device__ __forceinline__ unsigned short f32_to_bf16(float f) {
  union { float f; unsigned u; } x; x.f = f;
  unsigned u = x.u;
  u += 0x7fffu + ((u >> 16) & 1u);   // round-to-nearest-even
  return (unsigned short)(u >> 16);
}

// ASYNCcnt-tracked direct global->LDS copy (16B per lane), per CDNA5 ISA 10/15.18.
#define ASYNC_LD_B128(ldsoff, gptr)                                   \
  asm volatile("global_load_async_to_lds_b128 %0, %1, off"            \
               :: "v"(ldsoff), "v"(gptr) : "memory")
// Async loads complete in order: waiting ASYNCcnt<=5 guarantees the previous
// tile's 5 loads landed while the 5 just-issued stay in flight.
#define WAIT_ASYNC(n) asm volatile("s_wait_asynccnt " #n ::: "memory")

__global__ void cvt_f32_bf16(const float* __restrict__ src,
                             unsigned short* __restrict__ dst, int n) {
  int i = blockIdx.x * blockDim.x + threadIdx.x;
  if (i < n) dst[i] = f32_to_bf16(src[i]);
}

// dst[c*rows + r] = bf16(src[r*cols + c])   (builds K-contiguous B^T tiles)
__global__ void transpose_cvt(const float* __restrict__ src,
                              unsigned short* __restrict__ dst,
                              int rows, int cols) {
  int c = blockIdx.x * 32 + (threadIdx.x & 31);
  int r = blockIdx.y * 8 + (threadIdx.x >> 5);
  if (r < rows && c < cols)
    dst[(size_t)c * rows + r] = f32_to_bf16(src[(size_t)r * cols + c]);
}

// C[M,Ncol] (f32) = A[M,512] (bf16 row-major) * B (given as BT[Ncol,512] bf16 row-major)
// Block tile 64(M) x 256(N); 8 waves as 2(m-groups of 32) x 4(n-groups of 64).
// Each wave: 8 accumulators (2x4 subtiles) -> 12 ds_b128 per 8 WMMAs.
// Double-buffered LDS staging via global_load_async_to_lds_b128; K loop fully
// unrolled so the ping-pong buffer, addresses and wait thresholds are static.
__global__ __launch_bounds__(256)
void gemm_bf16_wmma(const unsigned short* __restrict__ Abf,
                    const unsigned short* __restrict__ BTbf,
                    float* __restrict__ C, int M, int Ncol) {
  constexpr int K    = 512;
  constexpr int LDA  = 40;  // padded LDS row stride (elems): 80B keeps b128 align,
                            // lane l hits bank l*20 mod 64 -> 16 distinct banks
  constexpr unsigned ABUF = 64 * LDA * 2;    // 5120 B per A buffer
  constexpr unsigned BROW = 64 * LDA * 2;    // 5120 B per 64-row B chunk
  constexpr unsigned BBUF = 256 * LDA * 2;   // 20480 B per B buffer
  __shared__ unsigned short lA[2][64 * LDA];
  __shared__ unsigned short lB[2][256 * LDA];

  const int t    = threadIdx.x;
  const int wave = t >> 5;
  const int lane = t & 31;
  const int wm   = wave & 1;    // m group (32 rows = 2 subtiles)
  const int wn   = wave >> 1;   // n group (64 cols = 4 subtiles)
  const int h    = lane >> 4;   // lane half per ISA 16-bit A/B layouts
  const int l16  = lane & 15;

  const int tm0 = blockIdx.y * 64;
  const int tn0 = blockIdx.x * 256;

  // cooperative staging coords: 16B chunks, 4 per 32-elem row
  const int cr = t >> 2, cq = (t & 3) << 3;  // A row cr; B rows cr + 64*c

  const unsigned short* gA = Abf + (size_t)(tm0 + cr) * K + cq;
  const unsigned short* gB[4];
#pragma unroll
  for (int c = 0; c < 4; ++c)
    gB[c] = BTbf + (size_t)(tn0 + cr + 64 * c) * K + cq;

  // LDS byte offsets (wave-relative = low 32 bits of generic shared pointer);
  // ping-pong buffers are contiguous, so buffer select is pure arithmetic.
  const unsigned lA0 = (unsigned)(size_t)(void*)&lA[0][cr * LDA + cq];
  const unsigned lB0 = (unsigned)(size_t)(void*)&lB[0][cr * LDA + cq];

  // prologue: stage tile 0 into buffer 0
  ASYNC_LD_B128(lA0, gA);
#pragma unroll
  for (int c = 0; c < 4; ++c) ASYNC_LD_B128(lB0 + c * BROW, gB[c]);

  v8f acc[2][4] = {{{}, {}, {}, {}}, {{}, {}, {}, {}}};

#pragma unroll
  for (int k0 = 0; k0 < K; k0 += 32) {
    const int buf = (k0 >> 5) & 1;  // compile-time after full unroll
    __syncthreads();  // all waves done reading buf^1 (two tiles ago) -> safe to refill
    if (k0 + 32 < K) {
      ASYNC_LD_B128(lA0 + (buf ^ 1) * ABUF, gA + k0 + 32);
#pragma unroll
      for (int c = 0; c < 4; ++c)
        ASYNC_LD_B128(lB0 + (buf ^ 1) * BBUF + c * BROW, gB[c] + k0 + 32);
      WAIT_ASYNC(0x5);  // current tile landed; next tile stays in flight
    } else {
      WAIT_ASYNC(0x0);
    }
    __syncthreads();  // every wave's loads for `buf` complete

    // Preload ALL fragments, then issue the 8 WMMAs back-to-back.
    // A 16x32 frag: e0..7 -> K = h*8 + e ; e8..15 -> K = 16 + h*8 + (e-8)
    v16bf afr[2];
#pragma unroll
    for (int s = 0; s < 2; ++s) {
      const unsigned short* arow = lA[buf] + (wm * 32 + s * 16 + l16) * LDA;
      v8bf lo = *(const v8bf*)(arow + h * 8);
      v8bf hi = *(const v8bf*)(arow + 16 + h * 8);
#pragma unroll
      for (int i = 0; i < 8; ++i) { afr[s][i] = lo[i]; afr[s][i + 8] = hi[i]; }
    }
    // B 32x16 frag: e0..7 -> K = h*8 + e ; e8..15 -> K = 16 + h*8 + (e-8)
    v16bf bfr[4];
#pragma unroll
    for (int j = 0; j < 4; ++j) {
      const unsigned short* brow = lB[buf] + (wn * 64 + j * 16 + l16) * LDA;
      v8bf lo = *(const v8bf*)(brow + h * 8);
      v8bf hi = *(const v8bf*)(brow + 16 + h * 8);
#pragma unroll
      for (int i = 0; i < 8; ++i) { bfr[j][i] = lo[i]; bfr[j][i + 8] = hi[i]; }
    }
#pragma unroll
    for (int s = 0; s < 2; ++s)
#pragma unroll
      for (int j = 0; j < 4; ++j)
        acc[s][j] = __builtin_amdgcn_wmma_f32_16x16x32_bf16(
            false, afr[s], false, bfr[j], (short)0, acc[s][j], false, false);
  }

  // C/D layout: VGPR v, lane -> (M = h*8 + v, N = l16) within each 16x16 tile
#pragma unroll
  for (int s = 0; s < 2; ++s)
#pragma unroll
    for (int j = 0; j < 4; ++j)
#pragma unroll
      for (int v = 0; v < 8; ++v)
        C[(size_t)(tm0 + wm * 32 + s * 16 + h * 8 + v) * Ncol +
          tn0 + wn * 64 + j * 16 + l16] = acc[s][j][v];
}

// scores[n,w] = X[n+w-W] . Y[n] (0 for padded slots, which ARE in the softmax),
// softmax over the 21-slot window. One wave per row n.
__global__ __launch_bounds__(256)
void attn_softmax(const float* __restrict__ X, const float* __restrict__ Y,
                  float* __restrict__ attn) {
  const int wave = threadIdx.x >> 5;
  const int lane = threadIdx.x & 31;
  const int n = blockIdx.x * 8 + wave;
  const int m = n + lane - W_ATT;
  const bool inwin = lane < WIN;
  float s = 0.f;
  if (inwin && m >= 0 && m < N_UTT) {
    const float* xr = X + (size_t)m * D_EMB;
    const float* yr = Y + (size_t)n * D_EMB;
    float acc = 0.f;
    for (int d = 0; d < D_EMB; ++d) acc += xr[d] * yr[d];
    s = acc;
  }
  float v = inwin ? s : -3.0e38f;
  float mx = v;
#pragma unroll
  for (int off = 16; off > 0; off >>= 1) mx = fmaxf(mx, __shfl_xor(mx, off, 32));
  float e = inwin ? __expf(v - mx) : 0.f;
  float sum = e;
#pragma unroll
  for (int off = 16; off > 0; off >>= 1) sum += __shfl_xor(sum, off, 32);
  if (inwin) attn[(size_t)n * WIN + lane] = e / sum;
}

// out[n,e] = relu( HW[n, 4*512+e]  +  sum_w attn[n,w]*(HW[m, r1*512+e] + HW[m, r2*512+e]) )
// r1 = pred/suc selector (m>=n), r2 = same/diff speaker selector. Band never materialized.
__global__ __launch_bounds__(256)
void banded_aggregate(const float* __restrict__ HW, const float* __restrict__ attn,
                      const int* __restrict__ spk,
                      float* __restrict__ outF, unsigned short* __restrict__ outB) {
  const int n = blockIdx.y;
  const int e = blockIdx.x * 256 + threadIdx.x;
  const int sn = spk[n];
  float acc = HW[(size_t)n * HWCOLS + 4 * D_EMB + e];  // h @ w_aggr term
#pragma unroll
  for (int w = 0; w < WIN; ++w) {
    const int m = n + w - W_ATT;
    if (m < 0 || m >= N_UTT) continue;
    const float a = attn[(size_t)n * WIN + w];
    const float* row = HW + (size_t)m * HWCOLS;
    const int r1 = (m >= n) ? 0 : 1;           // pred (j>=i) vs suc
    const int r2 = (spk[m] == sn) ? 2 : 3;     // same vs diff speaker
    acc += a * (row[r1 * D_EMB + e] + row[r2 * D_EMB + e]);
  }
  const float r = fmaxf(acc, 0.f);
  if (outF) outF[(size_t)n * D_EMB + e] = r;
  if (outB) outB[(size_t)n * D_EMB + e] = f32_to_bf16(r);
}

extern "C" void kernel_launch(void* const* d_in, const int* in_sizes, int n_in,
                              void* d_out, int out_size, void* d_ws, size_t ws_size,
                              hipStream_t stream) {
  const float* X     = (const float*)d_in[0];
  const int*   spk   = (const int*)d_in[1];
  const float* w_gc1 = (const float*)d_in[2];
  const float* w_gc2 = (const float*)d_in[3];
  const float* w_att = (const float*)d_in[4];
  const float* w_ag1 = (const float*)d_in[5];
  const float* w_ag2 = (const float*)d_in[6];
  float* out = (float*)d_out;

  char* p = (char*)d_ws;
  auto carve = [&](size_t bytes) -> void* {
    void* q = (void*)p;
    p += (bytes + 255) & ~(size_t)255;
    return q;
  };
  unsigned short* Xb    = (unsigned short*)carve((size_t)N_UTT * D_EMB * 2);
  unsigned short* WattT = (unsigned short*)carve((size_t)D_EMB * D_EMB * 2);
  unsigned short* Wc1   = (unsigned short*)carve((size_t)NREL * D_EMB * D_EMB * 2);
  unsigned short* Wc2   = (unsigned short*)carve((size_t)NREL * D_EMB * D_EMB * 2);
  unsigned short* H1b   = (unsigned short*)carve((size_t)N_UTT * D_EMB * 2);
  float* Y       = (float*)carve((size_t)N_UTT * D_EMB * 4);
  float* attnBuf = (float*)carve((size_t)N_UTT * WIN * 4);
  float* HW      = (float*)carve((size_t)N_UTT * HWCOLS * 4);  // reused by both layers

  // 1) operand conversion / weight packing (B^T, K-contiguous, bf16)
  cvt_f32_bf16<<<(N_UTT * D_EMB) / 256, 256, 0, stream>>>(X, Xb, N_UTT * D_EMB);
  dim3 tb(256), tg(D_EMB / 32, D_EMB / 8);
  transpose_cvt<<<tg, tb, 0, stream>>>(w_att, WattT, D_EMB, D_EMB);
  for (int r = 0; r < 4; ++r) {
    transpose_cvt<<<tg, tb, 0, stream>>>(w_gc1 + (size_t)r * D_EMB * D_EMB,
                                         Wc1 + (size_t)r * D_EMB * D_EMB, D_EMB, D_EMB);
    transpose_cvt<<<tg, tb, 0, stream>>>(w_gc2 + (size_t)r * D_EMB * D_EMB,
                                         Wc2 + (size_t)r * D_EMB * D_EMB, D_EMB, D_EMB);
  }
  transpose_cvt<<<tg, tb, 0, stream>>>(w_ag1, Wc1 + (size_t)4 * D_EMB * D_EMB, D_EMB, D_EMB);
  transpose_cvt<<<tg, tb, 0, stream>>>(w_ag2, Wc2 + (size_t)4 * D_EMB * D_EMB, D_EMB, D_EMB);

  // 2) Y = X @ W_att ;  attn = softmax(windowed scores)
  gemm_bf16_wmma<<<dim3(D_EMB / 256, N_UTT / 64), 256, 0, stream>>>(
      Xb, WattT, Y, N_UTT, D_EMB);
  attn_softmax<<<N_UTT / 8, 256, 0, stream>>>(X, Y, attnBuf);

  // 3) layer 1: HW = X @ [W_gc1(0..3) | W_aggr1] ; banded aggregate -> H1 (bf16)
  gemm_bf16_wmma<<<dim3(HWCOLS / 256, N_UTT / 64), 256, 0, stream>>>(
      Xb, Wc1, HW, N_UTT, HWCOLS);
  banded_aggregate<<<dim3(D_EMB / 256, N_UTT), 256, 0, stream>>>(
      HW, attnBuf, spk, nullptr, H1b);

  // 4) layer 2: HW = H1 @ [W_gc2(0..3) | W_aggr2] ; banded aggregate -> out (f32)
  gemm_bf16_wmma<<<dim3(HWCOLS / 256, N_UTT / 64), 256, 0, stream>>>(
      H1b, Wc2, HW, N_UTT, HWCOLS);
  banded_aggregate<<<dim3(D_EMB / 256, N_UTT), 256, 0, stream>>>(
      HW, attnBuf, spk, out, nullptr);
}
